// MetaQuickSR_46127948759103
// MI455X (gfx1250) — matrix operand order repaired and checked by reading, not compiled, not run on hardware
//
#include <hip/hip_runtime.h>
#include <stdint.h>

// ---------------------------------------------------------------------------
// MetaQuickSR on MI455X (gfx1250, wave32, WMMA + async-to-LDS)
//
//   convert_w2       : w2 (256x432 f32) -> f16 col-major, zero-padded to 448
//   conv3x3_wmma<CIN>: implicit-GEMM 3x3 conv. LDS-staged, pre-swizzled A/B
//                      fragments -> ds_load_b128 + v_wmma inner loop.
//   p2w_fused        : per 16-row tile of P=262144:
//                        feature patch -> LDS via global_load_async_to_lds_b32
//                        h  = relu(pos@w1+b1)  (VALU)       -> LDS f16
//                        lw = h@w2 + b2        (WMMA f16)   -> LDS f32
//                        s_wait_asynccnt 0; out = cols.lw + 255*mean -> HBM
//                      lw (453 MB if materialized) never leaves LDS; the
//                      async DMA overlaps the whole WMMA GEMM.
// ---------------------------------------------------------------------------

typedef __attribute__((ext_vector_type(16))) _Float16 v16h;
typedef __attribute__((ext_vector_type(8)))  _Float16 v8h;
typedef __attribute__((ext_vector_type(8)))  float    v8f;

union V16H { v16h v; v8h h[2]; _Float16 e[16]; };
union V8F  { v8f  v; float e[8]; };

#define HPITCH 264   // 256 + 8 halves
#define LWPITCH 440  // 432 + 8 floats
#define FPITCH 12    // 10 + 2 floats (feature patch rows)

// ---------------------------------------------------------------------------
__global__ __launch_bounds__(256) void convert_w2(const float* __restrict__ w2,
                                                  _Float16* __restrict__ w2T) {
  int idx = blockIdx.x * 256 + threadIdx.x;   // over 448*256
  if (idx < 448 * 256) {
    int col = idx >> 8;
    int k   = idx & 255;
    float v = (col < 432) ? w2[k * 432 + col] : 0.0f;
    w2T[col * 256 + k] = (_Float16)v;
  }
}

// ---------------------------------------------------------------------------
// 3x3 SAME conv + bias + ReLU, implicit GEMM. M=16 out-ch, K=CIN*9.
// Block covers 128 pixels of one image row; wave w -> pixels [16w,16w+16).
// grid.x = N * H * (W/128) = 2048.
// ---------------------------------------------------------------------------
template <int CIN>
__global__ __launch_bounds__(256)
void conv3x3_wmma(const float* __restrict__ fin, const float* __restrict__ wts,
                  const float* __restrict__ bias, float* __restrict__ fout) {
  constexpr int KTOT   = CIN * 9;              // 27 | 144
  constexpr int KSTEPS = (KTOT + 31) / 32;     // 1  | 5
  constexpr int KP     = KSTEPS * 32;          // 32 | 160
  constexpr int BP     = KP + 8;               // 16B-aligned col pitch (halves)

  __shared__ _Float16 bF[128 * BP];            // B fragments: [col][k]
  __shared__ _Float16 wF[KSTEPS * 32 * 16];    // A fragments: [ks][lane][i]

  const int t    = threadIdx.x;
  const int lane = t & 31;
  const int wave = t >> 5;
  const int blk  = blockIdx.x;
  const int wbase = (blk & 1) * 128;
  const int h     = (blk >> 1) & 255;
  const int n     = blk >> 9;
  const int half = lane >> 4;
  const int l16  = lane & 15;

  // ---- Stage A fragments (weights, pre-swizzled to WMMA lane order).
  // A layout: lane (m=l16, half), element i -> K = (i/8)*16 + half*8 + i%8.
  for (int idx = t; idx < KSTEPS * 32 * 16; idx += 256) {
    int i  = idx & 15;
    int ln = (idx >> 4) & 31;
    int ks = idx >> 9;
    int m  = ln & 15, hf = ln >> 4;
    int k  = ks * 32 + ((i >> 3) << 4) + hf * 8 + (i & 7);
    float v = 0.0f;
    if (k < KTOT) v = wts[(m * CIN + k / 9) * 9 + (k % 9)];
    wF[idx] = (_Float16)v;
  }

  // ---- Stage B: bF[col][k] = fin[ci][h+kh-1][wbase+col+kw-1] (zero-padded).
  // Unconditional clamped loads -> no exec-mask serialization.
  for (int idx = t; idx < KP * 128; idx += 256) {
    int col = idx & 127;
    int k   = idx >> 7;
    float v = 0.0f;
    if (k < KTOT) {
      int ci = k / 9, tap = k - ci * 9;
      int kh = tap / 3, kw = tap - kh * 3;
      int hy = h + kh - 1;
      int wx = wbase + col + kw - 1;
      int hyc = hy < 0 ? 0 : (hy > 255 ? 255 : hy);
      int wxc = wx < 0 ? 0 : (wx > 255 ? 255 : wx);
      float f = fin[((n * CIN + ci) * 256 + hyc) * 256 + wxc];
      v = (hy == hyc && wx == wxc) ? f : 0.0f;
    }
    bF[col * BP + k] = (_Float16)v;
  }
  __syncthreads();

  // ---- GEMM: pure ds_load_b128 + v_wmma.
  V8F acc;
#pragma unroll
  for (int j = 0; j < 8; ++j) acc.e[j] = 0.0f;

#pragma unroll
  for (int ks = 0; ks < KSTEPS; ++ks) {
    V16H a, b;
    const _Float16* pa = &wF[ks * 512 + lane * 16];
    a.h[0] = *(const v8h*)pa;
    a.h[1] = *(const v8h*)(pa + 8);
    const _Float16* pb = &bF[(wave * 16 + l16) * BP + ks * 32 + half * 8];
    b.h[0] = *(const v8h*)pb;
    b.h[1] = *(const v8h*)(pb + 16);
    acc.v = __builtin_amdgcn_wmma_f32_16x16x32_f16(
        false, a.v, false, b.v, (short)0, acc.v, false, false);
  }

  // D layout: VGPR j -> out-channel co = j + 8*half ; pixel col = l16.
#pragma unroll
  for (int j = 0; j < 8; ++j) {
    int co = j + 8 * half;
    float v = acc.e[j] + bias[co];
    v = v > 0.0f ? v : 0.0f;
    fout[((n * 16 + co) * 256 + h) * 256 + (wbase + wave * 16 + l16)] = v;
  }
}

// ---------------------------------------------------------------------------
// Fused Pos2Weight + locally-connected upsample.
// One block = one 16-row tile of the (P x 432) GEMM; rows share (hh, s1) and
// span 8 ww values, so the needed feature patch (4n x 16c x 3 x 10 f32) is
// DMA'd into LDS with async-to-LDS loads that overlap the WMMA GEMM.
// grid = 16384.
// ---------------------------------------------------------------------------
__global__ __launch_bounds__(256)
void p2w_fused(const float* __restrict__ pos, const float* __restrict__ w1,
               const float* __restrict__ b1, const _Float16* __restrict__ w2T,
               const float* __restrict__ b2, const float* __restrict__ feat,
               float* __restrict__ out) {
  __shared__ _Float16 hS[16 * HPITCH];    // h tile (A operand), f16
  __shared__ float    lwS[16 * LWPITCH];  // lw tile (never leaves LDS)
  __shared__ float    featS[192 * FPITCH];// feature patch, [n*16+ci][3][10]
  __shared__ float    posS[48];

  const int t    = threadIdx.x;
  const int lane = t & 31;
  const int wave = t >> 5;
  const int half = lane >> 4;
  const int l16  = lane & 15;
  const int rbase = blockIdx.x * 16;

  // Tile-constant position decode: r = hh*1024 + s1*512 + ww*2 + s2.
  const int s1  = (rbase >> 9) & 1;
  const int hh  = rbase >> 10;
  const int ww0 = (rbase >> 1) & 255;     // low 3 bits zero

  if (t < 48) posS[t] = pos[rbase * 3 + t];

  // ---- Stage feature patch. Interior tiles (uniform per block): per-lane
  // async DMA into LDS, tracked by ASYNCcnt; waited only just before use.
  const bool interior =
      (hh >= 1) && (hh <= 254) && (ww0 >= 8) && (ww0 <= 240);
  if (interior) {
    for (int idx = t; idx < 1920; idx += 256) {
      int col  = idx % 10;
      int rest = idx / 10;                // (n*16+ci)*3 + kr
      int kr   = rest % 3;
      uint32_t ldsOff =
          (uint32_t)(uintptr_t)&featS[rest * FPITCH + col];
      uint64_t gAddr = (uint64_t)(uintptr_t)
          &feat[((rest / 3) * 256 + (hh + kr - 1)) * 256 + (ww0 + col - 1)];
      asm volatile("global_load_async_to_lds_b32 %0, %1, off"
                   :: "v"(ldsOff), "v"(gAddr)
                   : "memory");
    }
  } else {
    for (int idx = t; idx < 1920; idx += 256) {
      int col  = idx % 10;
      int rest = idx / 10;
      int kr   = rest % 3;
      int hy = hh + kr - 1, wx = ww0 + col - 1;
      int hyc = hy < 0 ? 0 : (hy > 255 ? 255 : hy);
      int wxc = wx < 0 ? 0 : (wx > 255 ? 255 : wx);
      float f = feat[((rest / 3) * 256 + hyc) * 256 + wxc];
      featS[rest * FPITCH + col] = (hy == hyc && wx == wxc) ? f : 0.0f;
    }
  }
  __syncthreads();

  // ---- Phase A: h[16][256] = relu(pos @ w1 + b1); thread t owns column t.
  {
    const int col = t;
    const float wv0 = w1[col], wv1 = w1[256 + col], wv2 = w1[512 + col];
    const float bb  = b1[col];
#pragma unroll
    for (int r = 0; r < 16; ++r) {
      float hv = posS[r * 3] * wv0 + posS[r * 3 + 1] * wv1 +
                 posS[r * 3 + 2] * wv2 + bb;
      hv = hv > 0.0f ? hv : 0.0f;
      hS[r * HPITCH + col] = (_Float16)hv;
    }
  }
  __syncthreads();

  // ---- Phase B: lw = h @ w2 (+b2). Wave owns N-tiles wave + 8u (u<4);
  // tiles >= 27 are not stored (27 reads zero-padded w2T cols, 28-31 read past w2T).
  V8F acc[4];
#pragma unroll
  for (int u = 0; u < 4; ++u)
#pragma unroll
    for (int j = 0; j < 8; ++j) acc[u].e[j] = 0.0f;

  for (int ks = 0; ks < 8; ++ks) {
    const int k0 = ks * 32;
    V16H a;
    const _Float16* pa = &hS[l16 * HPITCH + k0 + half * 16];
    a.h[0] = *(const v8h*)pa;
    a.h[1] = *(const v8h*)(pa + 8);

#pragma unroll
    for (int u = 0; u < 4; ++u) {
      const int col = (wave + 8 * u) * 16 + l16;          // < 512; cols >= 448 lie past w2T
      const _Float16* pb = &w2T[col * 256 + k0 + half * 16];
      if (ks < 7) __builtin_prefetch(pb + 32, 0, 3);      // next k-step, near
      V16H b;
      b.h[0] = *(const v8h*)pb;
      b.h[1] = *(const v8h*)(pb + 8);
      acc[u].v = __builtin_amdgcn_wmma_f32_16x16x32_f16(
          false, a.v, false, b.v, (short)0, acc[u].v, false, false);
    }
  }

#pragma unroll
  for (int u = 0; u < 4; ++u) {
    const int tile = wave + 8 * u;
    if (tile < 27) {
      const int col = tile * 16 + l16;
      const float bb = b2[col];
#pragma unroll
      for (int j = 0; j < 8; ++j) {
        const int r = j + 8 * half;                       // D: M = j + 8*half
        lwS[r * LWPITCH + col] = acc[u].e[j] + bb;
      }
    }
  }

  // Feature-patch DMA must be complete (per wave) before the barrier that
  // publishes featS to the whole block.
  asm volatile("s_wait_asynccnt 0" ::: "memory");
  __syncthreads();

  // ---- Phase C: out = cols . lw + 255*mean. Pure LDS + FMA, no branches.
  if (t < 192) {
    const int c    = t % 3;
    const int task = t / 3;
    const int rl   = task & 15;
    const int n    = task >> 4;
    const int tc   = rl >> 1;             // ww - ww0
    const int s2   = rl & 1;

    const float* lwr = &lwS[rl * LWPITCH];
    float accv = 0.0f;
    for (int ci = 0; ci < 16; ++ci) {
      const float* fS = &featS[(n * 16 + ci) * 3 * FPITCH];
      const float* lw9 = &lwr[ci * 27 + c];
#pragma unroll
      for (int kh = 0; kh < 3; ++kh)
#pragma unroll
        for (int kw = 0; kw < 3; ++kw)
          accv += fS[kh * FPITCH + tc + kw] * lw9[(kh * 3 + kw) * 3];
    }
    const float mean = (c == 0) ? 0.4488f : (c == 1) ? 0.4371f : 0.4040f;
    accv += 255.0f * mean;
    out[((n * 3 + c) * 512 + (2 * hh + s1)) * 512 +
        (2 * (ww0 + tc) + s2)] = accv;
  }
}

// ---------------------------------------------------------------------------
extern "C" void kernel_launch(void* const* d_in, const int* in_sizes, int n_in,
                              void* d_out, int out_size, void* d_ws,
                              size_t ws_size, hipStream_t stream) {
  const float* x   = (const float*)d_in[0];
  const float* pos = (const float*)d_in[1];
  const float* c0w = (const float*)d_in[2];
  const float* c0b = (const float*)d_in[3];
  const float* c1w = (const float*)d_in[4];
  const float* c1b = (const float*)d_in[5];
  const float* c2w = (const float*)d_in[6];
  const float* c2b = (const float*)d_in[7];
  const float* c3w = (const float*)d_in[8];
  const float* c3b = (const float*)d_in[9];
  const float* w1  = (const float*)d_in[10];
  const float* b1  = (const float*)d_in[11];
  const float* w2  = (const float*)d_in[12];
  const float* b2  = (const float*)d_in[13];
  float* out = (float*)d_out;

  // Workspace: fA | fB (4*16*256*256 f32 each) | w2T (448*256 f16)
  const size_t FEAT = (size_t)4 * 16 * 256 * 256;
  float* fA = (float*)d_ws;
  float* fB = fA + FEAT;
  _Float16* w2T = (_Float16*)(fB + FEAT);

  convert_w2<<<(448 * 256 + 255) / 256, 256, 0, stream>>>(w2, w2T);
  conv3x3_wmma<3><<<2048, 256, 0, stream>>>(x,  c0w, c0b, fA);
  conv3x3_wmma<16><<<2048, 256, 0, stream>>>(fA, c1w, c1b, fB);
  conv3x3_wmma<16><<<2048, 256, 0, stream>>>(fB, c2w, c2b, fA);
  conv3x3_wmma<16><<<2048, 256, 0, stream>>>(fA, c3w, c3b, fB);
  p2w_fused<<<16384, 256, 0, stream>>>(pos, w1, b1, w2T, b2, fB, out);
}
